// QuantizedLinear_5875515261093
// MI455X (gfx1250) — compile-verified, matrix operand order checked by reading, not run on hardware
//
#include <hip/hip_runtime.h>

// BitNet ternary linear: out[M,N] = (x[M,K] @ W[N,K]^T) * scale[N] + bias[N]
// W arrives as int32 in {-1,0,1}. Fused int32->bf16 dequant on the
// global->LDS path; bf16 WMMA with f32 accumulation.

#define M_DIM 256
#define K_DIM 4096
#define N_DIM 11008

#define BM 128
#define BN 128
#define KT 64
#define PAD 8
#define KSTRIDE (KT + PAD)        // 72 bf16 = 144 B row stride (16B aligned)
#define NSTAGES (K_DIM / KT)      // 64

typedef __attribute__((ext_vector_type(4)))  __bf16 bf16x4;
typedef __attribute__((ext_vector_type(8)))  __bf16 bf16x8;
typedef __attribute__((ext_vector_type(16))) __bf16 bf16x16;
typedef __attribute__((ext_vector_type(8)))  float  f32x8;

union FragAB {
    bf16x16 v;
    struct { bf16x8 lo, hi; } h;
};

__global__ __launch_bounds__(256)
void bitnet_wmma_bf16_kernel(const float* __restrict__ x,
                             const int*   __restrict__ w,
                             const float* __restrict__ scale,
                             const float* __restrict__ bias,
                             float*       __restrict__ out)
{
    __shared__ __bf16 xs[2][BM][KSTRIDE];
    __shared__ __bf16 ws[2][BN][KSTRIDE];

    const int tid  = threadIdx.x;
    const int lane = tid & 31;
    const int wave = tid >> 5;     // 0..7
    const int wm   = wave & 1;     // M slab: 0..1 (64 rows each)
    const int wn   = wave >> 1;    // N slab: 0..3 (32 cols each)
    const int half = lane >> 4;    // K sub-chunk select (A/B bf16 layout)
    const int r    = lane & 15;    // row/col within 16x16 tile

    const int bm0 = blockIdx.y * BM;
    const int bn0 = blockIdx.x * BN;

    // Prefetch registers for the next K stage: 8 x 16B per thread per operand.
    float4 xreg[8];
    int4   wreg[8];

    auto load_stage_global = [&](int k0) {
#pragma unroll
        for (int i = 0; i < 8; ++i) {
            const int c   = i * 256 + tid;      // 2048 16B-chunks per tile
            const int row = c >> 4;             // 16 chunks per 64-elem row
            const int col = (c & 15) << 2;
            xreg[i] = *(const float4*)(x + (size_t)(bm0 + row) * K_DIM + k0 + col);
            wreg[i] = *(const int4*)  (w + (size_t)(bn0 + row) * K_DIM + k0 + col);
        }
    };

    auto store_stage_lds = [&](int buf) {
#pragma unroll
        for (int i = 0; i < 8; ++i) {
            const int c   = i * 256 + tid;
            const int row = c >> 4;
            const int col = (c & 15) << 2;
            bf16x4 xb = { (__bf16)xreg[i].x, (__bf16)xreg[i].y,
                          (__bf16)xreg[i].z, (__bf16)xreg[i].w };
            bf16x4 wb = { (__bf16)(float)wreg[i].x, (__bf16)(float)wreg[i].y,
                          (__bf16)(float)wreg[i].z, (__bf16)(float)wreg[i].w };
            *(bf16x4*)(&xs[buf][row][col]) = xb;   // ds_store_b64
            *(bf16x4*)(&ws[buf][row][col]) = wb;   // ds_store_b64
        }
    };

    f32x8 acc[4][2] = {};  // 8 x (16x16 f32) accumulators per wave

    // Prologue: stage 0 into LDS buffer 0.
    load_stage_global(0);
    store_stage_lds(0);
    __syncthreads();

    for (int s = 0; s < NSTAGES; ++s) {
        const int cur = s & 1;

        // Kick off next stage's HBM loads before doing matrix math.
        if (s + 1 < NSTAGES) load_stage_global((s + 1) * KT);

#pragma unroll
        for (int ks = 0; ks < 2; ++ks) {        // two 16x16x32 k-steps per stage
            const int kb = ks * 32;
            FragAB a[4], b[2];
#pragma unroll
            for (int mi = 0; mi < 4; ++mi) {    // A: 16x32 bf16 fragments
                const __bf16* p = &xs[cur][wm * 64 + mi * 16 + r][kb + half * 8];
                a[mi].h.lo = *(const bf16x8*)p;          // ds_load_b128 (K 0..7 / 8..15)
                a[mi].h.hi = *(const bf16x8*)(p + 16);   // ds_load_b128 (K 16..23 / 24..31)
            }
#pragma unroll
            for (int ni = 0; ni < 2; ++ni) {    // B: 32x16 bf16 fragments (W row = out col)
                const __bf16* p = &ws[cur][wn * 32 + ni * 16 + r][kb + half * 8];
                b[ni].h.lo = *(const bf16x8*)p;
                b[ni].h.hi = *(const bf16x8*)(p + 16);
            }
#pragma unroll
            for (int mi = 0; mi < 4; ++mi)
#pragma unroll
                for (int ni = 0; ni < 2; ++ni)
                    acc[mi][ni] = __builtin_amdgcn_wmma_f32_16x16x32_bf16(
                        false, a[mi].v, false, b[ni].v,
                        (short)0, acc[mi][ni], false, false);
        }

        if (s + 1 < NSTAGES) {
            __syncthreads();                    // all waves done reading buf[cur^1]
            store_stage_lds((s + 1) & 1);       // convert + stage next tile
            __syncthreads();                    // next tile visible to all waves
        }
    }

    // Epilogue: per-output-channel scale + bias, from the documented
    // 16x16 f32 C/D layout (VGPR v -> row v + 8*half, lane r -> col).
    const int m_base = bm0 + wm * 64;
    const int n_base = bn0 + wn * 32;
#pragma unroll
    for (int ni = 0; ni < 2; ++ni) {
        const int   n  = n_base + ni * 16 + r;
        const float sc = scale[n];
        const float bi = bias[n];
#pragma unroll
        for (int mi = 0; mi < 4; ++mi) {
#pragma unroll
            for (int v = 0; v < 8; ++v) {
                const int m = m_base + mi * 16 + half * 8 + v;
                out[(size_t)m * N_DIM + n] = acc[mi][ni][v] * sc + bi;
            }
        }
    }
}

extern "C" void kernel_launch(void* const* d_in, const int* in_sizes, int n_in,
                              void* d_out, int out_size, void* d_ws, size_t ws_size,
                              hipStream_t stream) {
    (void)in_sizes; (void)n_in; (void)d_ws; (void)ws_size; (void)out_size;
    const float* x  = (const float*)d_in[0];
    const int*   w  = (const int*)  d_in[1];
    const float* sc = (const float*)d_in[2];
    const float* bi = (const float*)d_in[3];
    float*       o  = (float*)d_out;

    dim3 grid(N_DIM / BN, M_DIM / BM);   // (86, 2)
    dim3 block(256);                     // 8 waves (wave32)
    bitnet_wmma_bf16_kernel<<<grid, block, 0, stream>>>(x, w, sc, bi, o);
}
